// CrossSymbolAttention_28913719837055
// MI455X (gfx1250) — matrix operand order repaired and checked by reading, not compiled, not run on hardware
//
#include <hip/hip_runtime.h>
#include <hip/hip_bf16.h>

typedef __bf16 bf16;
typedef __attribute__((ext_vector_type(16))) __bf16 v16bf;
typedef __attribute__((ext_vector_type(8)))  __bf16 bf16x8;
typedef __attribute__((ext_vector_type(8)))  float  v8f;
typedef __attribute__((ext_vector_type(4)))  float  f32x4;

#define WMMA_BF16(A, B, C) \
  __builtin_amdgcn_wmma_f32_16x16x32_bf16(false, (A), false, (B), (short)0, (C), false, false)

static constexpr int   Hh  = 8;
static constexpr int   HD  = 64;
static constexpr int   Dd  = 512;
static constexpr int   Bb  = 64;
static constexpr int   Tt  = 2048;
static constexpr int   Mrows = Bb * Tt;          // 131072
static constexpr float EPS   = 1.1920929e-07f;   // finfo(float32).eps

// ---------------- fragment loaders (per CDNA5 ISA 16-bit layouts) ----------------

// A-matrix 16x32 bf16: lane m = lane&15; element e: K = (e<8?0:16) + (lane>=16?8:0) + e%8
__device__ __forceinline__ v16bf load_fragA_bf16(const bf16* __restrict__ base, int row,
                                                 int ld, int k0, int lane) {
  const int half = lane >> 4;
  const bf16* p = base + (size_t)row * ld + k0 + half * 8;
  bf16x8 lo = *(const bf16x8*)(p);
  bf16x8 hi = *(const bf16x8*)(p + 16);
  v16bf f;
#pragma unroll
  for (int i = 0; i < 8; ++i) { f[i] = lo[i]; f[i + 8] = hi[i]; }
  return f;
}

// A-matrix fragment sourced from fp32 memory (convert to bf16 in-register)
__device__ __forceinline__ v16bf load_fragA_f32(const float* __restrict__ base, int row,
                                                int ld, int k0, int lane) {
  const int half = lane >> 4;
  const float* p = base + (size_t)row * ld + k0 + half * 8;
  f32x4 a0 = *(const f32x4*)(p);
  f32x4 a1 = *(const f32x4*)(p + 4);
  f32x4 b0 = *(const f32x4*)(p + 16);
  f32x4 b1 = *(const f32x4*)(p + 20);
  v16bf f;
#pragma unroll
  for (int i = 0; i < 4; ++i) {
    f[i]      = (bf16)a0[i];
    f[i + 4]  = (bf16)a1[i];
    f[i + 8]  = (bf16)b0[i];
    f[i + 12] = (bf16)b1[i];
  }
  return f;
}

// B-matrix 32x16 bf16: lane n = lane&15; element e: K = (e<8?0:16) + (lane>=16?8:0) + e%8
// Loaded from row-major W so that B[k][n] = Wrow_n[k]  (i.e. pass row = n's source row)
__device__ __forceinline__ v16bf load_fragB_bf16(const bf16* __restrict__ base, int row,
                                                 int ld, int k0, int lane) {
  const int half = lane >> 4;
  const bf16* p = base + (size_t)row * ld + k0 + half * 8;
  bf16x8 lo = *(const bf16x8*)(p);
  bf16x8 hi = *(const bf16x8*)(p + 16);
  v16bf f;
#pragma unroll
  for (int i = 0; i < 8; ++i) { f[i] = lo[i]; f[i + 8] = hi[i]; }
  return f;
}

__device__ __forceinline__ float half_reduce_sum(float v) {
#pragma unroll
  for (int m = 1; m <= 8; m <<= 1) v += __shfl_xor(v, m, 32);
  return v;
}
__device__ __forceinline__ float half_reduce_max(float v) {
#pragma unroll
  for (int m = 1; m <= 8; m <<= 1) v = fmaxf(v, __shfl_xor(v, m, 32));
  return v;
}

// ---------------- prep kernels ----------------

__global__ __launch_bounds__(256) void cvt_weights_kernel(
    const float* __restrict__ Wq, const float* __restrict__ Wk,
    const float* __restrict__ Wv, const float* __restrict__ Wo,
    bf16* __restrict__ Wf, bf16* __restrict__ Wob) {
  int i = blockIdx.x * 256 + threadIdx.x;
  if (i < Dd * Dd) {
    Wf[i]               = (bf16)Wq[i];
    Wf[Dd * Dd + i]     = (bf16)Wk[i];
    Wf[2 * Dd * Dd + i] = (bf16)Wv[i];
    Wob[i]              = (bf16)Wo[i];
  }
}

__global__ void build_mask_kernel(const unsigned char* __restrict__ gm,
                                  unsigned long long* __restrict__ mb) {
  int b = threadIdx.x;  // 64 threads
  unsigned long long m = 1ull << b;  // | eye
  for (int c = 0; c < Bb; ++c)
    if (gm[b * Bb + c] != 0) m |= (1ull << c);
  mb[b] = m;
}

// ---------------- fused QKV projection + per-head RMS norm ----------------
// grid = Mrows/16 blocks, 256 threads (8 waves). Wave w owns head w for q, k, v.
// Single K-loop: each x A-fragment is loaded/converted once and feeds 12 WMMAs.
// Outputs: qn/kn/vn as bf16 [t][h][b][hd] (contiguous 64x64 tiles per (t,h)).
__global__ __launch_bounds__(256) void qkv_kernel(
    const float* __restrict__ x, const bf16* __restrict__ Wf,
    bf16* __restrict__ qn, bf16* __restrict__ kn, bf16* __restrict__ vn) {
  const int lane   = threadIdx.x & 31;
  const int wave   = threadIdx.x >> 5;
  const int m0     = blockIdx.x * 16;
  const int rowA   = m0 + (lane & 15);
  const int half   = lane >> 4;
  const int col_lo = lane & 15;

  v8f z = {0.f, 0.f, 0.f, 0.f, 0.f, 0.f, 0.f, 0.f};
  v8f acc[3][4] = {{z, z, z, z}, {z, z, z, z}, {z, z, z, z}};

#pragma unroll 2
  for (int k0 = 0; k0 < Dd; k0 += 32) {
    v16bf a = load_fragA_f32(x, rowA, Dd, k0, lane);
#pragma unroll
    for (int job = 0; job < 3; ++job) {
      const int wrow = job * Dd + wave * HD;
#pragma unroll
      for (int n = 0; n < 4; ++n) {
        v16bf b = load_fragB_bf16(Wf, wrow + n * 16 + col_lo, Dd, k0, lane);
        acc[job][n] = WMMA_BF16(a, b, acc[job][n]);
      }
    }
  }

  // RMS norm over the 64-wide head for q and k (jobs 0, 1)
#pragma unroll
  for (int job = 0; job < 2; ++job) {
#pragma unroll
    for (int j = 0; j < 8; ++j) {
      float ss = 0.f;
#pragma unroll
      for (int n = 0; n < 4; ++n) ss += acc[job][n][j] * acc[job][n][j];
      ss = half_reduce_sum(ss);
      float r = rsqrtf(ss * (1.0f / 64.0f) + EPS);
#pragma unroll
      for (int n = 0; n < 4; ++n) acc[job][n][j] *= r;
    }
  }

#pragma unroll
  for (int job = 0; job < 3; ++job) {
    bf16* dst = (job == 0) ? qn : (job == 1) ? kn : vn;
#pragma unroll
    for (int j = 0; j < 8; ++j) {
      const int mrow = m0 + j + 8 * half;  // global row = b*T + t
      const int bsym = mrow >> 11;
      const int t    = mrow & (Tt - 1);
      const size_t rb = (((size_t)t * Hh + wave) * Bb + bsym) * HD;
#pragma unroll
      for (int n = 0; n < 4; ++n) dst[rb + n * 16 + col_lo] = (bf16)acc[job][n][j];
    }
  }
}

// ---------------- attention: one block per (t,h), 4 waves x 16-row strips ----------------
__global__ __launch_bounds__(128) void attn_kernel(
    const bf16* __restrict__ qn, const bf16* __restrict__ kn,
    const bf16* __restrict__ vn, const unsigned long long* __restrict__ maskbits,
    bf16* __restrict__ ybf) {
  __shared__ bf16 ldsVT[HD * Bb];      // V transposed: [hd][c]
  __shared__ bf16 ldsP[4 * 16 * Bb];   // per-wave softmax rows: [wave][row][c]

  const int tid  = threadIdx.x;
  const int lane = tid & 31;
  const int wave = tid >> 5;
  const int th   = blockIdx.x;
  const int t    = th >> 3;
  const int h    = th & 7;
  const bf16* Q = qn + (size_t)th * (Bb * HD);
  const bf16* K = kn + (size_t)th * (Bb * HD);
  const bf16* V = vn + (size_t)th * (Bb * HD);

  // cooperative transpose of V (64x64 bf16, [c][hd]) into LDS [hd][c]
  {
    const int c  = tid >> 1;
    const int ch = (tid & 1) * 32;
    const bf16* vr = V + c * HD + ch;
    bf16x8 r0 = *(const bf16x8*)(vr);
    bf16x8 r1 = *(const bf16x8*)(vr + 8);
    bf16x8 r2 = *(const bf16x8*)(vr + 16);
    bf16x8 r3 = *(const bf16x8*)(vr + 24);
#pragma unroll
    for (int i = 0; i < 8; ++i) {
      ldsVT[(ch + i) * Bb + c]      = r0[i];
      ldsVT[(ch + 8 + i) * Bb + c]  = r1[i];
      ldsVT[(ch + 16 + i) * Bb + c] = r2[i];
      ldsVT[(ch + 24 + i) * Bb + c] = r3[i];
    }
  }
  __syncthreads();

  const int m_lo   = wave * 16;
  const int half   = lane >> 4;
  const int col_lo = lane & 15;

  // S = Q * K^T  (strip 16 x 64)
  v8f z = {0.f, 0.f, 0.f, 0.f, 0.f, 0.f, 0.f, 0.f};
  v8f s[4] = {z, z, z, z};
#pragma unroll
  for (int k0 = 0; k0 < HD; k0 += 32) {
    v16bf a = load_fragA_bf16(Q, m_lo + col_lo, HD, k0, lane);
#pragma unroll
    for (int n = 0; n < 4; ++n) {
      v16bf b = load_fragB_bf16(K, n * 16 + col_lo, HD, k0, lane);
      s[n] = WMMA_BF16(a, b, s[n]);
    }
  }

  // scale + mask
  unsigned long long mb[8];
#pragma unroll
  for (int j = 0; j < 8; ++j) mb[j] = maskbits[m_lo + j + 8 * half];
#pragma unroll
  for (int n = 0; n < 4; ++n) {
    const int c = n * 16 + col_lo;
#pragma unroll
    for (int j = 0; j < 8; ++j) {
      float v = s[n][j] * 0.125f;  // 1/sqrt(64)
      s[n][j] = ((mb[j] >> c) & 1ull) ? v : -1e30f;
    }
  }

  // softmax per row, write P (bf16) into this wave's LDS strip
  bf16* Pw = ldsP + wave * (16 * Bb);
#pragma unroll
  for (int j = 0; j < 8; ++j) {
    float mx = fmaxf(fmaxf(s[0][j], s[1][j]), fmaxf(s[2][j], s[3][j]));
    mx = half_reduce_max(mx);
    float e0 = exp2f((s[0][j] - mx) * 1.4426950408889634f);
    float e1 = exp2f((s[1][j] - mx) * 1.4426950408889634f);
    float e2 = exp2f((s[2][j] - mx) * 1.4426950408889634f);
    float e3 = exp2f((s[3][j] - mx) * 1.4426950408889634f);
    float sum = half_reduce_sum(e0 + e1 + e2 + e3);
    float inv = 1.0f / sum;
    const int r = (j + 8 * half) * Bb + col_lo;
    Pw[r]      = (bf16)(e0 * inv);
    Pw[r + 16] = (bf16)(e1 * inv);
    Pw[r + 32] = (bf16)(e2 * inv);
    Pw[r + 48] = (bf16)(e3 * inv);
  }
  __syncthreads();

  // Y = P * V  (strip 16 x 64); A from LDS P, B from LDS V^T rows
  v8f y[4] = {z, z, z, z};
#pragma unroll
  for (int k0 = 0; k0 < Bb; k0 += 32) {
    v16bf a = load_fragA_bf16(Pw, col_lo, Bb, k0, lane);
#pragma unroll
    for (int n = 0; n < 4; ++n) {
      v16bf b = load_fragB_bf16(ldsVT, n * 16 + col_lo, Bb, k0, lane);
      y[n] = WMMA_BF16(a, b, y[n]);
    }
  }
  // store y as bf16 [b][t][d]
#pragma unroll
  for (int j = 0; j < 8; ++j) {
    const int brow = m_lo + j + 8 * half;
    const size_t ob = ((size_t)brow * Tt + t) * Dd + h * HD;
#pragma unroll
    for (int n = 0; n < 4; ++n) ybf[ob + n * 16 + col_lo] = (bf16)y[n][j];
  }
}

// ---------------- output projection: out = y @ Wo^T (fp32 out) ----------------
__global__ __launch_bounds__(256) void out_kernel(
    const bf16* __restrict__ ybf, const bf16* __restrict__ Wob,
    float* __restrict__ out) {
  const int lane   = threadIdx.x & 31;
  const int wave   = threadIdx.x >> 5;
  const int m0     = blockIdx.x * 16;
  const int rowA   = m0 + (lane & 15);
  const int wbase  = wave * 64;   // 8 waves cover N = 512
  const int half   = lane >> 4;
  const int col_lo = lane & 15;

  v8f z = {0.f, 0.f, 0.f, 0.f, 0.f, 0.f, 0.f, 0.f};
  v8f acc[4] = {z, z, z, z};
#pragma unroll 4
  for (int k0 = 0; k0 < Dd; k0 += 32) {
    v16bf a = load_fragA_bf16(ybf, rowA, Dd, k0, lane);
#pragma unroll
    for (int n = 0; n < 4; ++n) {
      v16bf b = load_fragB_bf16(Wob, wbase + n * 16 + col_lo, Dd, k0, lane);
      acc[n] = WMMA_BF16(a, b, acc[n]);
    }
  }
#pragma unroll
  for (int j = 0; j < 8; ++j) {
    const size_t ob = (size_t)(m0 + j + 8 * half) * Dd + wbase;
#pragma unroll
    for (int n = 0; n < 4; ++n) out[ob + n * 16 + col_lo] = acc[n][j];
  }
}

// ---------------- launch ----------------

extern "C" void kernel_launch(void* const* d_in, const int* in_sizes, int n_in,
                              void* d_out, int out_size, void* d_ws, size_t ws_size,
                              hipStream_t stream) {
  const float*         x  = (const float*)d_in[0];
  const unsigned char* gm = (const unsigned char*)d_in[1];  // jax bool -> 1 byte
  const float*         Wq = (const float*)d_in[2];
  const float*         Wk = (const float*)d_in[3];
  const float*         Wv = (const float*)d_in[4];
  const float*         Wo = (const float*)d_in[5];
  float*               out = (float*)d_out;

  char* ws = (char*)d_ws;
  const size_t SZ_QKV = (size_t)Tt * Hh * Bb * HD * sizeof(bf16);  // 128 MiB each
  bf16*               Wf  = (bf16*)(ws + 0);                        // 1536x512 bf16
  bf16*               Wob = (bf16*)(ws + 1572864);                  // 512x512 bf16
  unsigned long long* mb  = (unsigned long long*)(ws + 2097152);    // 64 x u64
  bf16* qn  = (bf16*)(ws + 4194304);
  bf16* kn  = (bf16*)(ws + 4194304 + SZ_QKV);
  bf16* vn  = (bf16*)(ws + 4194304 + 2 * SZ_QKV);
  bf16* ybf = (bf16*)(ws + 4194304 + 3 * SZ_QKV);

  cvt_weights_kernel<<<(Dd * Dd + 255) / 256, 256, 0, stream>>>(Wq, Wk, Wv, Wo, Wf, Wob);
  build_mask_kernel<<<1, 64, 0, stream>>>(gm, mb);
  qkv_kernel<<<Mrows / 16, 256, 0, stream>>>(x, Wf, qn, kn, vn);
  attn_kernel<<<Tt * Hh, 128, 0, stream>>>(qn, kn, vn, mb, ybf);
  out_kernel<<<Mrows / 16, 256, 0, stream>>>(ybf, Wob, out);
}
